// RNNwithNALU_16140487098504
// MI455X (gfx1250) — matrix operand order repaired and checked by reading, not compiled, not run on hardware
//
#include <hip/hip_runtime.h>

// ---------------------------------------------------------------------------
// 2-layer LSTM (H=51, in=1) + NALU head, B=1024, T=1000.
// Persistent-block design: 64 blocks x 128 threads (4 wave32 waves).
// Each block owns a 16-row batch tile; weights pre-swizzled into LDS as
// WMMA B-fragments (f16); h-state kept in LDS as f16 A-fragments; gates
// computed with v_wmma_f32_16x16x32_f16 (f32 accumulate).
// Round 3: use hardware v_tanh_f32 when the builtin exists (guarded by
// __has_builtin, rcp/exp fallback otherwise).
// ---------------------------------------------------------------------------

typedef __attribute__((ext_vector_type(16))) _Float16 v16h;
typedef __attribute__((ext_vector_type(8)))  float    v8f;

#define HH     51
#define FOURH  204
#define NTILE  13      // ceil(204/16) -> N padded to 208
#define KC1    2       // K chunks for layer1 (K=51 padded to 64)
#define KC2    4       // K chunks for layer2 (K=[h1|h2]=128 padded layout)
#define BTILE  16      // batch rows per block
#define TCH    64      // timestep staging chunk
#define EPSV   1e-5f

struct __align__(16) SharedMem {
    _Float16 w1frag[KC1 * NTILE * 32 * 16];   // 26.0 KB  B-fragments Whh1^T
    _Float16 w2frag[KC2 * NTILE * 32 * 16];   // 52.0 KB  B-fragments [Wih2^T;Whh2^T]
    _Float16 h1buf [BTILE * 64];              // h1 as f16, K-padded to 64
    _Float16 h12buf[BTILE * 128];             // [h1 | pad | h2 | pad] K=128
    float    gates [BTILE * 208];             // WMMA f32 output staging
    float    c1    [BTILE * HH];
    float    c2    [BTILE * HH];
    float    h2f   [BTILE * HH];              // f32 h2 for NALU
    float    bias1 [FOURH];
    float    bias2 [FOURH];
    float    wih1  [FOURH];                   // W_ih1 column (input size 1)
    float    gvec  [HH];
    float    wnaluv[HH];
    float    wnacv [HH];
    float    xbuf  [BTILE * TCH];
    float    outbuf[BTILE * TCH];
};

// sigmoid via hardware v_rcp_f32 (no IEEE-div expansion)
__device__ __forceinline__ float sigf(float x) {
    return __builtin_amdgcn_rcpf(1.0f + __expf(-x));
}
// tanh: single v_tanh_f32 if available, else 1 - 2/(exp(2x)+1) (rcp(inf)=0)
__device__ __forceinline__ float tanhfast(float x) {
#if __has_builtin(__builtin_amdgcn_tanhf)
    return __builtin_amdgcn_tanhf(x);
#else
    return 1.0f - 2.0f * __builtin_amdgcn_rcpf(__expf(2.0f * x) + 1.0f);
#endif
}

__device__ __forceinline__ v16h load_frag2(const _Float16* p0, const _Float16* p1) {
    union { uint4 q[2]; v16h h; } u;
    u.q[0] = *(const uint4*)p0;
    u.q[1] = *(const uint4*)p1;
    return u.h;
}

__global__ __launch_bounds__(128)
void RNNwithNALU_16140487098504_kernel(
    const float* __restrict__ x,      // [B,T]
    const float* __restrict__ W_ih1,  // [204,1]
    const float* __restrict__ W_hh1,  // [204,51]
    const float* __restrict__ b_ih1,  // [204]
    const float* __restrict__ b_hh1,  // [204]
    const float* __restrict__ W_ih2,  // [204,51]
    const float* __restrict__ W_hh2,  // [204,51]
    const float* __restrict__ b_ih2,  // [204]
    const float* __restrict__ b_hh2,  // [204]
    const float* __restrict__ G,      // [1,51]
    const float* __restrict__ W_nalu, // [1,51]
    const float* __restrict__ W_nac,  // [51,1]
    float* __restrict__ out,          // [B,T]
    int B, int T)
{
    __shared__ SharedMem smem;

    const int tid  = threadIdx.x;          // 0..127
    const int wav  = tid >> 5;             // wave id 0..3
    const int lane = tid & 31;
    const int lrow = lane & 15;            // 0..15
    const int lhi  = lane >> 4;            // 0 or 1
    const int b0   = blockIdx.x * BTILE;

    // ---------------- init: zero state, biases, swizzled weight fragments ----
    for (int i = tid; i < BTILE * 64;  i += 128) smem.h1buf[i]  = (_Float16)0.f;
    for (int i = tid; i < BTILE * 128; i += 128) smem.h12buf[i] = (_Float16)0.f;
    for (int i = tid; i < BTILE * HH;  i += 128) { smem.c1[i] = 0.f; smem.c2[i] = 0.f; }
    for (int i = tid; i < FOURH; i += 128) {
        smem.bias1[i] = b_ih1[i] + b_hh1[i];
        smem.bias2[i] = b_ih2[i] + b_hh2[i];
        smem.wih1[i]  = W_ih1[i];
    }
    for (int i = tid; i < HH; i += 128) {
        smem.gvec[i]   = G[i];
        smem.wnaluv[i] = W_nalu[i];
        smem.wnacv[i]  = W_nac[i];
    }
    // B-fragment layout: frag[((kc*NTILE+nt)*32 + lane)*16 + s]
    //   holds Wt[K = kc*32 + (lane/16)*16 + s][N = nt*16 + lane%16]
    for (int f = tid; f < KC1 * NTILE * 512; f += 128) {
        int s  = f & 15;  int r1 = f >> 4;
        int ln = r1 & 31; int r2 = r1 >> 5;
        int nt = r2 % NTILE, kc = r2 / NTILE;
        int K = kc * 32 + ((ln >> 4) << 4) + s;
        int N = nt * 16 + (ln & 15);
        float v = (K < HH && N < FOURH) ? W_hh1[N * HH + K] : 0.f;  // Whh1^T[K][N]
        smem.w1frag[f] = (_Float16)v;
    }
    for (int f = tid; f < KC2 * NTILE * 512; f += 128) {
        int s  = f & 15;  int r1 = f >> 4;
        int ln = r1 & 31; int r2 = r1 >> 5;
        int nt = r2 % NTILE, kc = r2 / NTILE;
        int K = kc * 32 + ((ln >> 4) << 4) + s;
        int N = nt * 16 + (ln & 15);
        float v = 0.f;
        if (N < FOURH) {
            if (K < HH)                     v = W_ih2[N * HH + K];          // h1 part
            else if (K >= 64 && K < 64+HH)  v = W_hh2[N * HH + (K - 64)];   // h2 part
        }
        smem.w2frag[f] = (_Float16)v;
    }
    __syncthreads();

    // ---------------- time loop, staged in chunks of TCH ---------------------
    for (int tc = 0; tc < T; tc += TCH) {
        const int cw = min(TCH, T - tc);
        // stage x chunk (coalesced per row); fast path for full chunks
        if (cw == TCH) {
            for (int idx = tid; idx < BTILE * TCH; idx += 128) {
                int r = idx >> 6, c = idx & (TCH - 1);
                smem.xbuf[idx] = x[(size_t)(b0 + r) * T + tc + c];
            }
        } else {
            for (int idx = tid; idx < BTILE * cw; idx += 128) {
                int r = idx / cw, c = idx - r * cw;
                smem.xbuf[r * TCH + c] = x[(size_t)(b0 + r) * T + tc + c];
            }
        }
        // prefetch next chunk (global_prefetch_b8)
        if (tc + TCH < T && tid < BTILE)
            __builtin_prefetch(&x[(size_t)(b0 + tid) * T + tc + TCH], 0, 0);
        __syncthreads();

        for (int tt = 0; tt < cw; ++tt) {
            // ---- Phase A: gates1 = h1 @ Whh1^T  (WMMA) ----
            for (int nt = wav; nt < NTILE; nt += 4) {
                v8f acc = {};
#pragma unroll
                for (int kc = 0; kc < KC1; ++kc) {
                    const _Float16* ap = &smem.h1buf[lrow * 64 + kc * 32 + lhi * 16];
                    v16h a = load_frag2(ap, ap + 8);
                    const _Float16* bp = &smem.w1frag[((kc * NTILE + nt) * 32 + lane) * 16];
                    v16h b = load_frag2(bp, bp + 8);
                    acc = __builtin_amdgcn_wmma_f32_16x16x32_f16(
                        false, a, false, b, (short)0, acc, false, false);
                }
                union { v8f v; float f[8]; } u; u.v = acc;
#pragma unroll
                for (int v = 0; v < 8; ++v)
                    smem.gates[(lhi * 8 + v) * 208 + nt * 16 + lrow] = u.f[v];
            }
            __syncthreads();

            // ---- Phase B: LSTM1 elementwise; writes h1 (f16) ----
            for (int idx = tid; idx < BTILE * HH; idx += 128) {
                int r = idx / HH, j = idx - r * HH;
                float xv = smem.xbuf[r * TCH + tt];
                float gi = smem.gates[r*208 +        j] + smem.bias1[       j] + xv * smem.wih1[       j];
                float gf = smem.gates[r*208 +   HH + j] + smem.bias1[  HH + j] + xv * smem.wih1[  HH + j];
                float gg = smem.gates[r*208 + 2*HH + j] + smem.bias1[2*HH + j] + xv * smem.wih1[2*HH + j];
                float go = smem.gates[r*208 + 3*HH + j] + smem.bias1[3*HH + j] + xv * smem.wih1[3*HH + j];
                float c = sigf(gf) * smem.c1[idx] + sigf(gi) * tanhfast(gg);
                smem.c1[idx] = c;
                float h = sigf(go) * tanhfast(c);
                smem.h1buf [r * 64  + j] = (_Float16)h;
                smem.h12buf[r * 128 + j] = (_Float16)h;
            }
            __syncthreads();

            // ---- Phase C: gates2 = [h1|h2] @ [Wih2^T;Whh2^T]  (WMMA) ----
            for (int nt = wav; nt < NTILE; nt += 4) {
                v8f acc = {};
#pragma unroll
                for (int kc = 0; kc < KC2; ++kc) {
                    const _Float16* ap = &smem.h12buf[lrow * 128 + kc * 32 + lhi * 16];
                    v16h a = load_frag2(ap, ap + 8);
                    const _Float16* bp = &smem.w2frag[((kc * NTILE + nt) * 32 + lane) * 16];
                    v16h b = load_frag2(bp, bp + 8);
                    acc = __builtin_amdgcn_wmma_f32_16x16x32_f16(
                        false, a, false, b, (short)0, acc, false, false);
                }
                union { v8f v; float f[8]; } u; u.v = acc;
#pragma unroll
                for (int v = 0; v < 8; ++v)
                    smem.gates[(lhi * 8 + v) * 208 + nt * 16 + lrow] = u.f[v];
            }
            __syncthreads();

            // ---- Phase D: LSTM2 elementwise; writes h2 (f16 + f32) ----
            for (int idx = tid; idx < BTILE * HH; idx += 128) {
                int r = idx / HH, j = idx - r * HH;
                float gi = smem.gates[r*208 +        j] + smem.bias2[       j];
                float gf = smem.gates[r*208 +   HH + j] + smem.bias2[  HH + j];
                float gg = smem.gates[r*208 + 2*HH + j] + smem.bias2[2*HH + j];
                float go = smem.gates[r*208 + 3*HH + j] + smem.bias2[3*HH + j];
                float c = sigf(gf) * smem.c2[idx] + sigf(gi) * tanhfast(gg);
                smem.c2[idx] = c;
                float h = sigf(go) * tanhfast(c);
                smem.h12buf[r * 128 + 64 + j] = (_Float16)h;
                smem.h2f[idx] = h;
            }
            __syncthreads();

            // ---- Phase E: NALU head; 8 lanes per row, shuffle-reduce ----
            {
                int r = tid >> 3, p = tid & 7;
                float nac = 0.f, gd = 0.f, ml = 0.f;
                for (int j = p; j < HH; j += 8) {
                    float h = smem.h2f[r * HH + j];
                    nac += h * smem.wnacv[j];
                    gd  += h * smem.gvec[j];
                    ml  += __logf(fabsf(h) + EPSV) * smem.wnaluv[j];
                }
#pragma unroll
                for (int m = 1; m <= 4; m <<= 1) {
                    nac += __shfl_xor(nac, m, 32);
                    gd  += __shfl_xor(gd,  m, 32);
                    ml  += __shfl_xor(ml,  m, 32);
                }
                if (p == 0) {
                    float g = sigf(gd);
                    smem.outbuf[r * TCH + tt] = g * nac + (1.f - g) * __expf(ml);
                }
            }
            // no barrier needed here: next write to gates (Phase A) is fenced
            // by the barrier after Phase D; outbuf flush below has its own.
        }
        __syncthreads();
        // flush output chunk (coalesced per row); fast path for full chunks
        if (cw == TCH) {
            for (int idx = tid; idx < BTILE * TCH; idx += 128) {
                int r = idx >> 6, c = idx & (TCH - 1);
                out[(size_t)(b0 + r) * T + tc + c] = smem.outbuf[idx];
            }
        } else {
            for (int idx = tid; idx < BTILE * cw; idx += 128) {
                int r = idx / cw, c = idx - r * cw;
                out[(size_t)(b0 + r) * T + tc + c] = smem.outbuf[r * TCH + c];
            }
        }
        __syncthreads();
    }
}

extern "C" void kernel_launch(void* const* d_in, const int* in_sizes, int n_in,
                              void* d_out, int out_size, void* d_ws, size_t ws_size,
                              hipStream_t stream) {
    const float* x      = (const float*)d_in[0];
    const float* W_ih1  = (const float*)d_in[1];
    const float* W_hh1  = (const float*)d_in[2];
    const float* b_ih1  = (const float*)d_in[3];
    const float* b_hh1  = (const float*)d_in[4];
    const float* W_ih2  = (const float*)d_in[5];
    const float* W_hh2  = (const float*)d_in[6];
    const float* b_ih2  = (const float*)d_in[7];
    const float* b_hh2  = (const float*)d_in[8];
    const float* G      = (const float*)d_in[9];
    const float* W_nalu = (const float*)d_in[10];
    const float* W_nac  = (const float*)d_in[11];
    float* out = (float*)d_out;

    const int B = 1024;
    const int T = in_sizes[0] / B;   // 1000

    dim3 grid(B / BTILE);            // 64 persistent blocks
    dim3 block(128);                 // 4 wave32 waves
    RNNwithNALU_16140487098504_kernel<<<grid, block, 0, stream>>>(
        x, W_ih1, W_hh1, b_ih1, b_hh1, W_ih2, W_hh2, b_ih2, b_hh2,
        G, W_nalu, W_nac, out, B, T);
}
